// EncoderBlock_2783138808316
// MI455X (gfx1250) — matrix operand order repaired and checked by reading, not compiled, not run on hardware
//
#include <hip/hip_runtime.h>
#include <hip/hip_bf16.h>

typedef __attribute__((ext_vector_type(16))) _Float16 v16h;
typedef __attribute__((ext_vector_type(8)))  _Float16 v8h;
typedef __attribute__((ext_vector_type(8)))  float    v8f;
typedef __attribute__((address_space(3)))    void     lds_void;

#define Bb   32
#define Ss   577
#define Dd   768
#define Hh   12
#define Dh   64
#define FFf  3072
#define MROWS (Bb * Ss)   // 18464

__device__ __forceinline__ v16h join8(v8h a, v8h b) {
    v16h r;
#pragma unroll
    for (int i = 0; i < 8; ++i) { r[i] = a[i]; r[i + 8] = b[i]; }
    return r;
}

__device__ __forceinline__ v8f wmma_f16(v16h a, v16h b, v8f c) {
    // D = A(16x32 f16) * B(32x16 f16) + C(16x16 f32)
    return __builtin_amdgcn_wmma_f32_16x16x32_f16(false, a, false, b, (short)0, c,
                                                  false, false);
}

// 32-bit LDS byte offset of a shared-memory pointer (generic -> AS3 addrspacecast)
__device__ __forceinline__ unsigned lds_addr(void* p) {
    return (unsigned)(unsigned long long)(lds_void*)p;
}

// CDNA5 async global->LDS copy (16B per lane), GVS addressing: saddr(u64)+vaddr(u32).
// Tracked by ASYNCcnt; consume after s_wait_asynccnt 0 + barrier.
__device__ __forceinline__ void async_ld_b128(unsigned ldsOff, const void* sbase,
                                              unsigned vOff) {
    asm volatile("global_load_async_to_lds_b128 %0, %1, %2"
                 :
                 : "v"(ldsOff), "v"(vOff), "s"(sbase)
                 : "memory");
}
__device__ __forceinline__ void wait_async0() {
    asm volatile("s_wait_asynccnt 0x0" ::: "memory");
}

// ---------------------------------------------------------------------------
// Weight packing: f32 -> f16
// ---------------------------------------------------------------------------
__global__ void cvt_f16_kernel(const float* __restrict__ in, _Float16* __restrict__ out, int n) {
    int i = blockIdx.x * 256 + threadIdx.x;
    if (i < n) out[i] = (_Float16)in[i];
}

// Wq/Wk/Wv [H][D][Dh] f32 -> Wqkv f16 [D][3*768], cols: [0,768)=Q [768,1536)=K [1536,2304)=V
__global__ void pack_qkv_kernel(const float* __restrict__ Wq, const float* __restrict__ Wk,
                                const float* __restrict__ Wv, _Float16* __restrict__ out) {
    int i = blockIdx.x * 256 + threadIdx.x;
    const int total = Dd * 3 * Dd;
    if (i >= total) return;
    int d = i / (3 * Dd), c = i % (3 * Dd);
    int sel = c / Dd, rem = c % Dd, h = rem >> 6, e = rem & 63;
    const float* W = (sel == 0) ? Wq : (sel == 1) ? Wk : Wv;
    out[i] = (_Float16)W[((size_t)h * Dd + d) * Dh + e];
}

// ---------------------------------------------------------------------------
// LayerNorm: f32 [rows][768] -> f16 [rows][768]
// ---------------------------------------------------------------------------
__global__ __launch_bounds__(256) void ln_kernel(const float* __restrict__ in,
                                                 const float* __restrict__ w,
                                                 const float* __restrict__ b,
                                                 _Float16* __restrict__ out, int rows) {
    int row = blockIdx.x;
    if (row >= rows) return;
    const float* p = in + (size_t)row * Dd;
    float vals[3], s = 0.f, ss = 0.f;
#pragma unroll
    for (int i = 0; i < 3; ++i) {
        float v = p[threadIdx.x + i * 256];
        vals[i] = v; s += v; ss += v * v;
    }
#pragma unroll
    for (int off = 16; off; off >>= 1) {
        s  += __shfl_down(s, off, 32);
        ss += __shfl_down(ss, off, 32);
    }
    __shared__ float r0[8], r1[8];
    int wid = threadIdx.x >> 5, lid = threadIdx.x & 31;
    if (lid == 0) { r0[wid] = s; r1[wid] = ss; }
    __syncthreads();
    if (wid == 0) {
        float a = (lid < 8) ? r0[lid] : 0.f;
        float c = (lid < 8) ? r1[lid] : 0.f;
#pragma unroll
        for (int off = 4; off; off >>= 1) {
            a += __shfl_down(a, off, 32);
            c += __shfl_down(c, off, 32);
        }
        if (lid == 0) { r0[0] = a; r1[0] = c; }
    }
    __syncthreads();
    float mu  = r0[0] * (1.f / Dd);
    float var = r1[0] * (1.f / Dd) - mu * mu;
    float rs  = rsqrtf(var + 1e-5f);
#pragma unroll
    for (int i = 0; i < 3; ++i) {
        int c = threadIdx.x + i * 256;
        out[(size_t)row * Dd + c] = (_Float16)((vals[i] - mu) * rs * w[c] + b[c]);
    }
}

// ---------------------------------------------------------------------------
// WMMA GEMM: C[M,N] = A[M,K](f16) * B[K,N](f16).
// Macro tile 128x128, K-step 32, 256 threads = 8 waves, wave tile 64x32.
// Double-buffered LDS; A staged with async global->LDS (ASYNCcnt), B staged
// via register prefetch + in-LDS transpose.
// EPI 0: QKV split-write (bias, q scaled 1/8)    -> q/k/v [B,H,S,Dh] f16
// EPI 1: + bias[c] + resid[r,c]                  -> f32 out
// EPI 2: gelu(acc + bias[c])                     -> f16 out
// ---------------------------------------------------------------------------
template <int EPI>
__global__ __launch_bounds__(256) void gemm_wmma(
    const _Float16* __restrict__ A, const _Float16* __restrict__ Bm,
    int M, int N, int K,
    const float* __restrict__ bias, const float* __restrict__ resid,
    float* __restrict__ outF, _Float16* __restrict__ outH,
    _Float16* __restrict__ q, _Float16* __restrict__ k, _Float16* __restrict__ v,
    const float* __restrict__ bq, const float* __restrict__ bk, const float* __restrict__ bv) {
    __shared__ _Float16 As[2][128][40];  // [m][k], padded
    __shared__ _Float16 Bs[2][128][40];  // [n][k] (transposed), padded

    const int t = threadIdx.x;
    const int mBase = blockIdx.y * 128;
    const int nBase = blockIdx.x * 128;
    const int wave = t >> 5, lane = t & 31;
    const int wm = (wave & 1) * 64, wn = (wave >> 1) * 32;
    const int hi = lane >> 4, ln = lane & 15;

    // A tile 128x32: 512 16B segments, 2 per thread, async copy-through
    auto issueA = [&](int kk, int buf) {
#pragma unroll
        for (int i = 0; i < 2; ++i) {
            int idx = t + i * 256;
            int r = idx >> 2, sg = (idx & 3) * 8;
            int gr = mBase + r;
            if (gr >= M) gr = M - 1;  // rows >= M are discarded at store
            async_ld_b128(lds_addr(&As[buf][r][sg]), A,
                          (unsigned)(((size_t)gr * K + kk + sg) * sizeof(_Float16)));
        }
    };
    // B tile 32x128 row-major: 512 16B segments, 2 per thread -> registers
    auto loadBreg = [&](int kk, v8h* reg) {
#pragma unroll
        for (int i = 0; i < 2; ++i) {
            int idx = t + i * 256;
            int kr = idx >> 4, nsg = (idx & 15) * 8;
            reg[i] = *(const v8h*)(Bm + (size_t)(kk + kr) * N + nBase + nsg);
        }
    };
    auto storeB = [&](int buf, const v8h* reg) {
#pragma unroll
        for (int i = 0; i < 2; ++i) {
            int idx = t + i * 256;
            int kr = idx >> 4, nsg = (idx & 15) * 8;
#pragma unroll
            for (int j = 0; j < 8; ++j) Bs[buf][nsg + j][kr] = reg[i][j];
        }
    };

    v8f acc[4][2] = {};
    v8h breg[2];

    // prologue: stage tile 0
    issueA(0, 0);
    loadBreg(0, breg);
    storeB(0, breg);
    wait_async0();
    __syncthreads();

    int cur = 0;
    for (int kk = 0; kk < K; kk += 32) {
        const int nxt = cur ^ 1;
        const bool hasNext = (kk + 32) < K;
        if (hasNext) {            // overlap next-tile staging with this tile's WMMAs
            issueA(kk + 32, nxt);
            loadBreg(kk + 32, breg);
        }
        v16h afr[4], bfr[2];
#pragma unroll
        for (int s = 0; s < 4; ++s) {
            const _Float16* ap = &As[cur][wm + s * 16 + ln][0];
            afr[s] = join8(*(const v8h*)(ap + 8 * hi), *(const v8h*)(ap + 8 * hi + 16));
        }
#pragma unroll
        for (int s = 0; s < 2; ++s) {
            const _Float16* bp = &Bs[cur][wn + s * 16 + ln][0];
            bfr[s] = join8(*(const v8h*)(bp + 8 * hi), *(const v8h*)(bp + 8 * hi + 16));
        }
#pragma unroll
        for (int i = 0; i < 4; ++i)
#pragma unroll
            for (int j = 0; j < 2; ++j) acc[i][j] = wmma_f16(afr[i], bfr[j], acc[i][j]);
        if (hasNext) storeB(nxt, breg);
        wait_async0();
        __syncthreads();
        cur = nxt;
    }

    // Epilogue. C layout: VGPR r -> row (r + 8*hi), lanes -> col (ln)
#pragma unroll
    for (int i = 0; i < 4; ++i)
#pragma unroll
        for (int j = 0; j < 2; ++j)
#pragma unroll
            for (int r = 0; r < 8; ++r) {
                int gr = mBase + wm + i * 16 + r + 8 * hi;
                int gc = nBase + wn + j * 16 + ln;
                if (gr >= M) continue;
                float val = acc[i][j][r];
                if constexpr (EPI == 0) {
                    int sel = gc / Dd, rem = gc % Dd, h = rem >> 6, e = rem & 63;
                    const float* bb = (sel == 0) ? bq : (sel == 1) ? bk : bv;
                    val += bb[h * Dh + e];
                    if (sel == 0) val *= 0.125f;  // 1/sqrt(Dh)
                    int bi = gr / Ss, si = gr % Ss;
                    _Float16* dst = (sel == 0) ? q : (sel == 1) ? k : v;
                    dst[(((size_t)(bi * Hh + h)) * Ss + si) * Dh + e] = (_Float16)val;
                } else if constexpr (EPI == 1) {
                    val += bias[gc] + resid[(size_t)gr * N + gc];
                    outF[(size_t)gr * N + gc] = val;
                } else {
                    val += bias[gc];
                    val = 0.5f * val * (1.f + erff(val * 0.70710678118654752f));
                    outH[(size_t)gr * N + gc] = (_Float16)val;
                }
            }
}

// ---------------------------------------------------------------------------
// Flash attention. Grid (5, H, B), 256 threads. Workgroup = 128 query rows of
// one (b,h); 8 waves each own a 16-row Q tile; streams K/V in 32-key tiles.
// Q/K staged via async global->LDS; V transposed through registers.
// q is pre-scaled by 1/sqrt(Dh). Inputs f16 [B,H,S,Dh]; ctx f16 [B,S,D].
// ---------------------------------------------------------------------------
__global__ __launch_bounds__(256) void flash_attn(const _Float16* __restrict__ q,
                                                  const _Float16* __restrict__ k,
                                                  const _Float16* __restrict__ v,
                                                  _Float16* __restrict__ ctx) {
    __shared__ _Float16 Qs[128][72];
    __shared__ _Float16 Ks[32][72];       // [key][e]
    __shared__ _Float16 Vt[64][40];       // [e][key] (transposed)
    __shared__ _Float16 Ps[8][16][32];    // per-wave P tile

    const int qblk = blockIdx.x, h = blockIdx.y, b = blockIdx.z;
    const size_t head = ((size_t)(b * Hh + h)) * Ss;
    const _Float16* qp = q + head * Dh;
    const _Float16* kp = k + head * Dh;
    const _Float16* vp = v + head * Dh;

    const int t = threadIdx.x;
    const int qBase = qblk * 128;
    const int wave = t >> 5, lane = t & 31;
    const int hi = lane >> 4, ln = lane & 15;

    // async-stage Q tile 128x64 (clamp OOB rows; stores are guarded later)
#pragma unroll
    for (int i = 0; i < 4; ++i) {
        int idx = t + i * 256;
        int r = idx >> 3, sg = (idx & 7) * 8;
        int gs = qBase + r; if (gs >= Ss) gs = Ss - 1;
        async_ld_b128(lds_addr(&Qs[r][sg]), qp,
                      (unsigned)(((size_t)gs * Dh + sg) * sizeof(_Float16)));
    }
    wait_async0();
    __syncthreads();

    v16h qf[2];
#pragma unroll
    for (int e = 0; e < 2; ++e) {
        const _Float16* p0 = &Qs[wave * 16 + ln][e * 32];
        qf[e] = join8(*(const v8h*)(p0 + 8 * hi), *(const v8h*)(p0 + 8 * hi + 16));
    }

    v8f O[4] = {};
    float mrow[8], lrow[8];
#pragma unroll
    for (int r = 0; r < 8; ++r) { mrow[r] = -1e30f; lrow[r] = 0.f; }

    const int nkt = (Ss + 31) / 32;  // 19
    for (int kt = 0; kt < nkt; ++kt) {
        const int kBase = kt * 32;
        {   // K tile 32x64: async copy-through (layout matches)
            int r = t >> 3, sg = (t & 7) * 8;
            int gs = kBase + r; if (gs >= Ss) gs = Ss - 1;
            async_ld_b128(lds_addr(&Ks[r][sg]), kp,
                          (unsigned)(((size_t)gs * Dh + sg) * sizeof(_Float16)));
        }
        {   // V tile 32x64 -> transposed Vt[e][key]
            int r = t >> 3, sg = (t & 7) * 8;
            int gs = kBase + r;
            v8h val = {};
            if (gs < Ss) val = *(const v8h*)(vp + (size_t)gs * Dh + sg);
#pragma unroll
            for (int j = 0; j < 8; ++j) Vt[sg + j][r] = val[j];
        }
        wait_async0();
        __syncthreads();

        // preload all 4 K fragments, then issue 4 WMMAs back-to-back
        v16h kb[2][2];
#pragma unroll
        for (int e = 0; e < 2; ++e) {
            const _Float16* kr0 = &Ks[ln][e * 32 + 8 * hi];
            const _Float16* kr1 = &Ks[16 + ln][e * 32 + 8 * hi];
            kb[0][e] = join8(*(const v8h*)kr0, *(const v8h*)(kr0 + 16));
            kb[1][e] = join8(*(const v8h*)kr1, *(const v8h*)(kr1 + 16));
        }
        v8f s0 = {}, s1 = {};
        s0 = wmma_f16(qf[0], kb[0][0], s0);
        s1 = wmma_f16(qf[0], kb[1][0], s1);
        s0 = wmma_f16(qf[1], kb[0][1], s0);
        s1 = wmma_f16(qf[1], kb[1][1], s1);

        // mask invalid keys (last tile)
        const float NEG = -1e30f;
        if (kBase + ln >= Ss) {
#pragma unroll
            for (int r = 0; r < 8; ++r) s0[r] = NEG;
        }
        if (kBase + 16 + ln >= Ss) {
#pragma unroll
            for (int r = 0; r < 8; ++r) s1[r] = NEG;
        }
        // online softmax per row (rows live in 16-lane halves)
#pragma unroll
        for (int r = 0; r < 8; ++r) {
            float mv = fmaxf(s0[r], s1[r]);
#pragma unroll
            for (int off = 1; off < 16; off <<= 1) mv = fmaxf(mv, __shfl_xor(mv, off, 32));
            float nm = fmaxf(mrow[r], mv);
            float p0 = __expf(s0[r] - nm);
            float p1 = __expf(s1[r] - nm);
            float rs = p0 + p1;
#pragma unroll
            for (int off = 1; off < 16; off <<= 1) rs += __shfl_xor(rs, off, 32);
            float alpha = __expf(mrow[r] - nm);
            lrow[r] = lrow[r] * alpha + rs;
            mrow[r] = nm;
            O[0][r] *= alpha; O[1][r] *= alpha; O[2][r] *= alpha; O[3][r] *= alpha;
            Ps[wave][r + 8 * hi][ln]      = (_Float16)p0;
            Ps[wave][r + 8 * hi][16 + ln] = (_Float16)p1;
        }
        // intra-wave LDS RAW: make P writes visible before fragment reload
        asm volatile("s_wait_dscnt 0x0" ::: "memory");

        const _Float16* pp = &Ps[wave][ln][0];
        v16h pf = join8(*(const v8h*)(pp + 8 * hi), *(const v8h*)(pp + 8 * hi + 16));
        v16h vb[4];
#pragma unroll
        for (int n = 0; n < 4; ++n) {
            const _Float16* vv = &Vt[n * 16 + ln][8 * hi];
            vb[n] = join8(*(const v8h*)vv, *(const v8h*)(vv + 16));
        }
#pragma unroll
        for (int n = 0; n < 4; ++n) O[n] = wmma_f16(pf, vb[n], O[n]);
        __syncthreads();
    }

    // write ctx [B,S,D] f16
    const int qr = qBase + wave * 16;
#pragma unroll
    for (int n = 0; n < 4; ++n)
#pragma unroll
        for (int r = 0; r < 8; ++r) {
            int si = qr + r + 8 * hi;
            if (si < Ss)
                ctx[((size_t)(b * Ss + si)) * Dd + h * Dh + n * 16 + ln] =
                    (_Float16)(O[n][r] / lrow[r]);
        }
}

// ---------------------------------------------------------------------------
extern "C" void kernel_launch(void* const* d_in, const int* in_sizes, int n_in,
                              void* d_out, int out_size, void* d_ws, size_t ws_size,
                              hipStream_t stream) {
    (void)in_sizes; (void)n_in; (void)out_size; (void)ws_size;
    const float* x    = (const float*)d_in[0];
    const float* ln1w = (const float*)d_in[1];
    const float* ln1b = (const float*)d_in[2];
    const float* Wq   = (const float*)d_in[3];
    const float* bq   = (const float*)d_in[4];
    const float* Wk   = (const float*)d_in[5];
    const float* bk   = (const float*)d_in[6];
    const float* Wv   = (const float*)d_in[7];
    const float* bv   = (const float*)d_in[8];
    const float* Wo   = (const float*)d_in[9];
    const float* bo   = (const float*)d_in[10];
    const float* ln2w = (const float*)d_in[11];
    const float* ln2b = (const float*)d_in[12];
    const float* W1   = (const float*)d_in[13];
    const float* b1   = (const float*)d_in[14];
    const float* W2   = (const float*)d_in[15];
    const float* b2   = (const float*)d_in[16];

    // workspace carving (~213 MB total)
    char* base = (char*)d_ws;
    size_t off = 0;
    auto carve = [&](size_t bytes) -> char* {
        char* p = base + off;
        off += (bytes + 255) & ~(size_t)255;
        return p;
    };
    const size_t headElems = (size_t)Bb * Hh * Ss * Dh;     // 14,180,352
    _Float16* Wqkv16 = (_Float16*)carve((size_t)Dd * 3 * Dd * 2);
    _Float16* Wo16   = (_Float16*)carve((size_t)Dd * Dd * 2);
    _Float16* W116   = (_Float16*)carve((size_t)Dd * FFf * 2);
    _Float16* W216   = (_Float16*)carve((size_t)FFf * Dd * 2);
    _Float16* hBuf   = (_Float16*)carve((size_t)MROWS * Dd * 2);   // LN out (reused)
    float*    x2     = (float*)   carve((size_t)MROWS * Dd * 4);   // post-attn residual
    char*     qkvRegion = base + off;
    _Float16* qB   = (_Float16*)carve(headElems * 2);
    _Float16* kB   = (_Float16*)carve(headElems * 2);
    _Float16* vB   = (_Float16*)carve(headElems * 2);
    _Float16* ctxB = (_Float16*)carve((size_t)MROWS * Dd * 2);
    _Float16* hidden = (_Float16*)qkvRegion;  // aliases dead q/k/v/ctx (113.4 MB)

    // 1) weights -> f16
    pack_qkv_kernel<<<(Dd * 3 * Dd + 255) / 256, 256, 0, stream>>>(Wq, Wk, Wv, Wqkv16);
    cvt_f16_kernel<<<(Dd * Dd + 255) / 256, 256, 0, stream>>>(Wo, Wo16, Dd * Dd);
    cvt_f16_kernel<<<(Dd * FFf + 255) / 256, 256, 0, stream>>>(W1, W116, Dd * FFf);
    cvt_f16_kernel<<<(FFf * Dd + 255) / 256, 256, 0, stream>>>(W2, W216, FFf * Dd);

    const dim3 blk(256);
    // 2) LN1
    ln_kernel<<<MROWS, blk, 0, stream>>>(x, ln1w, ln1b, hBuf, MROWS);
    // 3) QKV projection
    gemm_wmma<0><<<dim3((3 * Dd) / 128, (MROWS + 127) / 128), blk, 0, stream>>>(
        hBuf, Wqkv16, MROWS, 3 * Dd, Dd, nullptr, nullptr, nullptr, nullptr,
        qB, kB, vB, bq, bk, bv);
    // 4) attention -> ctx
    flash_attn<<<dim3((Ss + 127) / 128, Hh, Bb), blk, 0, stream>>>(qB, kB, vB, ctxB);
    // 5) Wo + bo + x residual -> x2 (f32)
    gemm_wmma<1><<<dim3(Dd / 128, (MROWS + 127) / 128), blk, 0, stream>>>(
        ctxB, Wo16, MROWS, Dd, Dd, bo, x, x2, nullptr,
        nullptr, nullptr, nullptr, nullptr, nullptr, nullptr);
    // 6) LN2
    ln_kernel<<<MROWS, blk, 0, stream>>>(x2, ln2w, ln2b, hBuf, MROWS);
    // 7) W1 + b1 + GELU -> hidden (f16)
    gemm_wmma<2><<<dim3(FFf / 128, (MROWS + 127) / 128), blk, 0, stream>>>(
        hBuf, W116, MROWS, FFf, Dd, b1, nullptr, nullptr, hidden,
        nullptr, nullptr, nullptr, nullptr, nullptr, nullptr);
    // 8) W2 + b2 + x2 residual -> out (f32)
    gemm_wmma<1><<<dim3(Dd / 128, (MROWS + 127) / 128), blk, 0, stream>>>(
        hidden, W216, MROWS, Dd, FFf, b2, x2, (float*)d_out, nullptr,
        nullptr, nullptr, nullptr, nullptr, nullptr, nullptr);
}
